// AlignmentEncoder_2456721293463
// MI455X (gfx1250) — matrix operand order repaired and checked by reading, not compiled, not run on hardware
//
#include <hip/hip_runtime.h>

#define B_    16
#define T1_   2048
#define T2_   512
#define TEMP  0.0005f

typedef _Float16 v16h __attribute__((ext_vector_type(16)));
typedef _Float16 v8h  __attribute__((ext_vector_type(8)));
typedef float    v8f  __attribute__((ext_vector_type(8)));

__device__ __forceinline__ v16h cat16(v8h lo, v8h hi) {
  return __builtin_shufflevector(lo, hi, 0,1,2,3,4,5,6,7,8,9,10,11,12,13,14,15);
}
__device__ __forceinline__ v8f wmma16(v16h a, v16h b, v8f c) {
  return __builtin_amdgcn_wmma_f32_16x16x32_f16(false, a, false, b, (short)0, c,
                                                false, false);
}
// two contiguous 16B chunks -> A/B fragment halves for this lane
__device__ __forceinline__ v16h frag_ld(const _Float16* p) {
  return cat16(*(const v8h*)p, *(const v8h*)(p + 8));
}

// ---------------------------------------------------------------------------
// Prep kernels
// ---------------------------------------------------------------------------
// keys (B,T2,256) f32 -> (B,T2+2,256) f16 with zero guard rows at both ends
__global__ __launch_bounds__(256) void pad_keys_kernel(
    const float* __restrict__ in, _Float16* __restrict__ out) {
  const int i = blockIdx.x * 256 + threadIdx.x;
  if (i >= B_ * (T2_ + 2) * 256) return;
  const int ci = i & 255;
  const int rw = (i >> 8) % (T2_ + 2);
  const int b  = (i >> 8) / (T2_ + 2);
  out[i] = (rw >= 1 && rw <= T2_)
               ? (_Float16)in[((size_t)b * T2_ + rw - 1) * 256 + ci]
               : (_Float16)0.f;
}

// queries (B,80,T1) channel-major -> (B,T1+2,96) time-major f16,
// ch 80..95 = 0, zero guard rows at both ends
__global__ __launch_bounds__(256) void pad_qin_kernel(
    const float* __restrict__ q, _Float16* __restrict__ out) {
  const int i = blockIdx.x * 256 + threadIdx.x;
  if (i >= B_ * (T1_ + 2) * 96) return;
  const int ci = i % 96;
  const int rw = (i / 96) % (T1_ + 2);
  const int b  = (i / 96) / (T1_ + 2);
  float v = 0.f;
  if (ci < 80 && rw >= 1 && rw <= T1_)
    v = q[((size_t)b * 80 + ci) * T1_ + (rw - 1)];
  out[i] = (_Float16)v;
}

// weights [k][n] (k = dt*secvalid + ci, row stride Nvalid) ->
// f16 [n][Kpad] with per-section padding (secpad) and N padding (Nout>=Nvalid)
__global__ __launch_bounds__(256) void pack_w_kernel(
    const float* __restrict__ in, _Float16* __restrict__ out,
    int Nout, int Nvalid, int Kpad, int secpad, int secvalid) {
  const int i = blockIdx.x * 256 + threadIdx.x;
  if (i >= Nout * Kpad) return;
  const int n = i / Kpad, kp = i % Kpad;
  const int dt = kp / secpad, ci = kp % secpad;
  float v = 0.f;
  if (ci < secvalid && n < Nvalid) v = in[(dt * secvalid + ci) * Nvalid + n];
  out[i] = (_Float16)v;
}

// |row|^2 over 256 f16 channels
__global__ __launch_bounds__(256) void rownorm_kernel(
    const _Float16* __restrict__ x, float* __restrict__ out, int nrows) {
  const int row = blockIdx.x * 256 + threadIdx.x;
  if (row >= nrows) return;
  const _Float16* p = x + (size_t)row * 256;
  float s = 0.f;
  for (int i = 0; i < 32; ++i) {
    const v8h v = *(const v8h*)(p + i * 8);
#pragma unroll
    for (int e = 0; e < 8; ++e) { const float f = (float)v[e]; s = fmaf(f, f, s); }
  }
  out[row] = s;
}

// ---------------------------------------------------------------------------
// Key encoder, WMMA. Block = 32 timesteps of one batch. 8 waves.
// Stage1: H(32x512) = relu(im2col(keys,k3)(32x768) x W1 + b1)   (24 k-steps)
// Stage2: ke(32x256) = H x W2 + b2                               (16 k-steps)
// A-side im2col is a contiguous walk through the guard-padded keys16 buffer.
// ---------------------------------------------------------------------------
__global__ __launch_bounds__(256) void ke_enc_wmma(
    const _Float16* __restrict__ keys16,
    const _Float16* __restrict__ w1k, const float* __restrict__ kb1,
    const _Float16* __restrict__ w2k, const float* __restrict__ kb2,
    _Float16* __restrict__ ke16)
{
  const int tid = threadIdx.x, lane = tid & 31, wave = tid >> 5;
  const int half = lane >> 4, r = lane & 15;
  const int blocksPerB = T2_ / 32;
  const int b  = blockIdx.x / blocksPerB;
  const int tb = (blockIdx.x % blocksPerB) * 32;

  __shared__ _Float16 H[32][512];   // 32 KB
  const int mt = wave >> 2;         // 0..1

  { // ---- stage 1 ----
    const int ntb = (wave & 3) * 8;     // 8 n-tiles per wave
    v8f acc[8];
#pragma unroll
    for (int j = 0; j < 8; ++j)
#pragma unroll
      for (int e = 0; e < 8; ++e) acc[j][e] = 0.f;

    const int trow = tb + mt * 16 + r;  // 0-based t; padded row index = trow+dt
    const _Float16* Ab = keys16 + ((size_t)b * (T2_ + 2) + trow) * 256 + half * 16;
    for (int ks = 0; ks < 24; ++ks) {
      const int kb = ks * 32;
      const v16h a = frag_ld(Ab + kb);
      v16h bf[8];
#pragma unroll
      for (int j = 0; j < 8; ++j)
        bf[j] = frag_ld(w1k + (size_t)((ntb + j) * 16 + r) * 768 + kb + half * 16);
#pragma unroll
      for (int j = 0; j < 8; ++j) acc[j] = wmma16(a, bf[j], acc[j]);
    }
#pragma unroll
    for (int j = 0; j < 8; ++j) {
      const int col = (ntb + j) * 16 + r;
      const float bias = kb1[col];
#pragma unroll
      for (int rr = 0; rr < 8; ++rr)
        H[mt * 16 + half * 8 + rr][col] = (_Float16)fmaxf(acc[j][rr] + bias, 0.f);
    }
  }
  __syncthreads();

  { // ---- stage 2 ----
    const int ntb = (wave & 3) * 4;
    v8f acc[4];
#pragma unroll
    for (int j = 0; j < 4; ++j)
#pragma unroll
      for (int e = 0; e < 8; ++e) acc[j][e] = 0.f;

    const int lrow = mt * 16 + r;
    for (int ks = 0; ks < 16; ++ks) {
      const int kb = ks * 32;
      const v16h a = frag_ld(&H[lrow][kb + half * 16]);
      v16h bf[4];
#pragma unroll
      for (int j = 0; j < 4; ++j)
        bf[j] = frag_ld(w2k + (size_t)((ntb + j) * 16 + r) * 512 + kb + half * 16);
#pragma unroll
      for (int j = 0; j < 4; ++j) acc[j] = wmma16(a, bf[j], acc[j]);
    }
#pragma unroll
    for (int j = 0; j < 4; ++j) {
      const int col = (ntb + j) * 16 + r;
      const float bias = kb2[col];
#pragma unroll
      for (int rr = 0; rr < 8; ++rr) {
        const int t = tb + mt * 16 + half * 8 + rr;
        ke16[((size_t)b * T2_ + t) * 256 + col] = (_Float16)(acc[j][rr] + bias);
      }
    }
  }
}

// ---------------------------------------------------------------------------
// Query encoder, WMMA. Block = 64 timesteps. 8 waves.
// S1: H1(64x160)=relu(A(64x288) x W1q)   S2: H2(64x96)=relu(H1 x W2q)
// S3: qe(64x256)=H2 x W3q + b3
// ---------------------------------------------------------------------------
__global__ __launch_bounds__(256) void qe_enc_wmma(
    const _Float16* __restrict__ qin16,
    const _Float16* __restrict__ w1q, const float* __restrict__ qb1,
    const _Float16* __restrict__ w2q, const float* __restrict__ qb2,
    const _Float16* __restrict__ w3q, const float* __restrict__ qb3,
    _Float16* __restrict__ qe16)
{
  const int tid = threadIdx.x, lane = tid & 31, wave = tid >> 5;
  const int half = lane >> 4, r = lane & 15;
  const int blocksPerB = T1_ / 64;
  const int b  = blockIdx.x / blocksPerB;
  const int tb = (blockIdx.x % blocksPerB) * 64;

  __shared__ _Float16 H1[64][160];  // 20 KB
  __shared__ _Float16 H2[64][96];   // 12 KB

  const int mt = wave >> 1;         // 0..3 (same for all stages)

  { // ---- stage 1: K=288 (3 sections of 96), N=160 ----
    const int ntb = (wave & 1) * 5;
    v8f acc[5];
#pragma unroll
    for (int j = 0; j < 5; ++j)
#pragma unroll
      for (int e = 0; e < 8; ++e) acc[j][e] = 0.f;

    const int trow = tb + mt * 16 + r;
    const _Float16* Ab = qin16 + ((size_t)b * (T1_ + 2) + trow) * 96 + half * 16;
    for (int ks = 0; ks < 9; ++ks) {
      const int kb = ks * 32;
      const v16h a = frag_ld(Ab + kb);
      v16h bf[5];
#pragma unroll
      for (int j = 0; j < 5; ++j)
        bf[j] = frag_ld(w1q + (size_t)((ntb + j) * 16 + r) * 288 + kb + half * 16);
#pragma unroll
      for (int j = 0; j < 5; ++j) acc[j] = wmma16(a, bf[j], acc[j]);
    }
#pragma unroll
    for (int j = 0; j < 5; ++j) {
      const int col = (ntb + j) * 16 + r;
      const float bias = qb1[col];
#pragma unroll
      for (int rr = 0; rr < 8; ++rr)
        H1[mt * 16 + half * 8 + rr][col] = (_Float16)fmaxf(acc[j][rr] + bias, 0.f);
    }
  }
  __syncthreads();

  { // ---- stage 2: K=160, N=96 (cols 80..95 forced to 0) ----
    const int ntb = (wave & 1) * 3;
    v8f acc[3];
#pragma unroll
    for (int j = 0; j < 3; ++j)
#pragma unroll
      for (int e = 0; e < 8; ++e) acc[j][e] = 0.f;

    const int lrow = mt * 16 + r;
    for (int ks = 0; ks < 5; ++ks) {
      const int kb = ks * 32;
      const v16h a = frag_ld(&H1[lrow][kb + half * 16]);
      v16h bf[3];
#pragma unroll
      for (int j = 0; j < 3; ++j)
        bf[j] = frag_ld(w2q + (size_t)((ntb + j) * 16 + r) * 160 + kb + half * 16);
#pragma unroll
      for (int j = 0; j < 3; ++j) acc[j] = wmma16(a, bf[j], acc[j]);
    }
#pragma unroll
    for (int j = 0; j < 3; ++j) {
      const int col = (ntb + j) * 16 + r;
      const float bias = (col < 80) ? qb2[col] : 0.f;
#pragma unroll
      for (int rr = 0; rr < 8; ++rr)
        H2[mt * 16 + half * 8 + rr][col] = (_Float16)fmaxf(acc[j][rr] + bias, 0.f);
    }
  }
  __syncthreads();

  { // ---- stage 3: K=96, N=256, no relu ----
    const int ntb = (wave & 1) * 8;
    v8f acc[8];
#pragma unroll
    for (int j = 0; j < 8; ++j)
#pragma unroll
      for (int e = 0; e < 8; ++e) acc[j][e] = 0.f;

    const int lrow = mt * 16 + r;
    for (int ks = 0; ks < 3; ++ks) {
      const int kb = ks * 32;
      const v16h a = frag_ld(&H2[lrow][kb + half * 16]);
      v16h bf[8];
#pragma unroll
      for (int j = 0; j < 8; ++j)
        bf[j] = frag_ld(w3q + (size_t)((ntb + j) * 16 + r) * 96 + kb + half * 16);
#pragma unroll
      for (int j = 0; j < 8; ++j) acc[j] = wmma16(a, bf[j], acc[j]);
    }
#pragma unroll
    for (int j = 0; j < 8; ++j) {
      const int col = (ntb + j) * 16 + r;
      const float bias = qb3[col];
#pragma unroll
      for (int rr = 0; rr < 8; ++rr) {
        const int t = tb + mt * 16 + half * 8 + rr;
        qe16[((size_t)b * T1_ + t) * 256 + col] = (_Float16)(acc[j][rr] + bias);
      }
    }
  }
}

// ---------------------------------------------------------------------------
// Fused WMMA GEMM (16 x 512 x 256 per block) + log_softmax + prior +
// masked softmax.
// ---------------------------------------------------------------------------
__global__ __launch_bounds__(256) void attn_kernel(
    const _Float16* __restrict__ qe16, const _Float16* __restrict__ ke16,
    const float* __restrict__ q2, const float* __restrict__ k2,
    const float* __restrict__ prior, const unsigned char* __restrict__ mask,
    float* __restrict__ attn, float* __restrict__ attn_lp)
{
  const int tid  = threadIdx.x;
  const int lane = tid & 31;
  const int wave = tid >> 5;
  const int blocksPerB = T1_ / 16;
  const int b  = blockIdx.x / blocksPerB;
  const int m0 = (blockIdx.x % blocksPerB) * 16;

  __shared__ float S[16][T2_];
  __shared__ float q2s[16];
  __shared__ float red[16][16];

  if (tid < 16) q2s[tid] = q2[(size_t)b * T1_ + m0 + tid];
  __syncthreads();

  const int half = lane >> 4, r = lane & 15;
  const _Float16* Abase = qe16 + ((size_t)b * T1_ + m0 + r) * 256 + half * 16;
  const _Float16* Bbase = ke16 + ((size_t)b * T2_ + r) * 256 + half * 16;

  v8f acc[4];
#pragma unroll
  for (int j = 0; j < 4; ++j)
#pragma unroll
    for (int e = 0; e < 8; ++e) acc[j][e] = 0.f;

#pragma unroll
  for (int ks = 0; ks < 8; ++ks) {
    const int kb = ks * 32;
    const v16h a = frag_ld(Abase + kb);
    v16h bf[4];
#pragma unroll
    for (int j = 0; j < 4; ++j)
      bf[j] = frag_ld(Bbase + (size_t)(wave * 4 + j) * 16 * 256 + kb);
#pragma unroll
    for (int j = 0; j < 4; ++j) acc[j] = wmma16(a, bf[j], acc[j]);
  }

#pragma unroll
  for (int j = 0; j < 4; ++j) {
    const int col = (wave * 4 + j) * 16 + r;
    const float k2v = k2[(size_t)b * T2_ + col];
#pragma unroll
    for (int rr = 0; rr < 8; ++rr) {
      const int row = rr + half * 8;
      S[row][col] = -TEMP * (q2s[row] + k2v - 2.f * acc[j][rr]);
    }
  }
  __syncthreads();

  const int row = tid >> 4, sub = tid & 15;
  const int t1g = m0 + row;
  const size_t obase = ((size_t)b * T1_ + t1g) * T2_;
  const size_t pbase = (size_t)b * T2_ * T1_;

  float pm = -__builtin_inff();
  for (int j = 0; j < 32; ++j) pm = fmaxf(pm, S[row][sub + 16 * j]);
  red[row][sub] = pm;
  __syncthreads();
  float m = -__builtin_inff();
  for (int i = 0; i < 16; ++i) m = fmaxf(m, red[row][i]);
  __syncthreads();
  float ps = 0.f;
  for (int j = 0; j < 32; ++j) ps += __expf(S[row][sub + 16 * j] - m);
  red[row][sub] = ps;
  __syncthreads();
  float den = 0.f;
  for (int i = 0; i < 16; ++i) den += red[row][i];
  const float logden = __logf(den);
  __syncthreads();

  float pm2 = -__builtin_inff();
  for (int j = 0; j < 32; ++j) {
    const int col  = sub + 16 * j;
    const float lp = S[row][col] - m - logden;
    const float pr = prior[pbase + (size_t)col * T1_ + t1g];
    const float alp = lp + __logf(pr + 1e-8f);
    attn_lp[obase + col] = alp;
    const float mv = mask[b * T2_ + col] ? -__builtin_inff() : alp;
    S[row][col] = mv;
    pm2 = fmaxf(pm2, mv);
  }
  red[row][sub] = pm2;
  __syncthreads();
  float m2 = -__builtin_inff();
  for (int i = 0; i < 16; ++i) m2 = fmaxf(m2, red[row][i]);
  __syncthreads();
  float ps2 = 0.f;
  for (int j = 0; j < 32; ++j) ps2 += __expf(S[row][sub + 16 * j] - m2);
  red[row][sub] = ps2;
  __syncthreads();
  float den2 = 0.f;
  for (int i = 0; i < 16; ++i) den2 += red[row][i];
  const float inv = 1.f / den2;
  for (int j = 0; j < 32; ++j) {
    const int col = sub + 16 * j;
    attn[obase + col] = __expf(S[row][col] - m2) * inv;
  }
}

// ---------------------------------------------------------------------------
extern "C" void kernel_launch(void* const* d_in, const int* in_sizes, int n_in,
                              void* d_out, int out_size, void* d_ws, size_t ws_size,
                              hipStream_t stream) {
  (void)in_sizes; (void)n_in; (void)out_size; (void)ws_size;
  const float* queries = (const float*)d_in[0];
  const float* keys    = (const float*)d_in[1];
  const unsigned char* mask = (const unsigned char*)d_in[2];
  const float* prior   = (const float*)d_in[3];
  const float* kw1 = (const float*)d_in[4];
  const float* kb1 = (const float*)d_in[5];
  const float* kw2 = (const float*)d_in[6];
  const float* kb2 = (const float*)d_in[7];
  const float* qw1 = (const float*)d_in[8];
  const float* qb1 = (const float*)d_in[9];
  const float* qw2 = (const float*)d_in[10];
  const float* qb2 = (const float*)d_in[11];
  const float* qw3 = (const float*)d_in[12];
  const float* qb3 = (const float*)d_in[13];

  float* out = (float*)d_out;
  float* attn_out = out;
  float* attn_lp  = out + (size_t)B_ * T1_ * T2_;

  char* ws = (char*)d_ws;
  const size_t MB = 1024 * 1024;
  _Float16* keys16 = (_Float16*)(ws);                 // (B,T2+2,256) ~4.1 MB
  _Float16* qin16  = (_Float16*)(ws + 5  * MB);       // (B,T1+2,96) ~6.1 MB
  _Float16* ke16   = (_Float16*)(ws + 12 * MB);       //  4 MB
  _Float16* qe16   = (_Float16*)(ws + 16 * MB);       // 16 MB
  float*    k2     = (float*)   (ws + 32 * MB);               // 32 KB
  float*    q2     = (float*)   (ws + 32 * MB + 256 * 1024);  // 128 KB
  _Float16* w1k    = (_Float16*)(ws + 33 * MB);       // 512x768
  _Float16* w2k    = (_Float16*)(ws + 34 * MB);       // 256x512
  _Float16* w1q    = (_Float16*)(ws + 35 * MB);       // 160x288
  _Float16* w2q    = (_Float16*)(ws + 35 * MB + 256 * 1024); // 96x160
  _Float16* w3q    = (_Float16*)(ws + 35 * MB + 512 * 1024); // 256x96

  // prep
  pad_keys_kernel<<<(B_ * (T2_ + 2) * 256 + 255) / 256, 256, 0, stream>>>(
      keys, keys16);
  pad_qin_kernel<<<(B_ * (T1_ + 2) * 96 + 255) / 256, 256, 0, stream>>>(
      queries, qin16);
  pack_w_kernel<<<(512 * 768 + 255) / 256, 256, 0, stream>>>(
      kw1, w1k, 512, 512, 768, 256, 256);
  pack_w_kernel<<<(256 * 512 + 255) / 256, 256, 0, stream>>>(
      kw2, w2k, 256, 256, 512, 512, 512);
  pack_w_kernel<<<(160 * 288 + 255) / 256, 256, 0, stream>>>(
      qw1, w1q, 160, 160, 288, 96, 80);
  pack_w_kernel<<<(96 * 160 + 255) / 256, 256, 0, stream>>>(
      qw2, w2q, 96, 80, 160, 160, 160);
  pack_w_kernel<<<(256 * 96 + 255) / 256, 256, 0, stream>>>(
      qw3, w3q, 256, 256, 96, 96, 80);

  // encoders (WMMA)
  ke_enc_wmma<<<B_ * (T2_ / 32), 256, 0, stream>>>(
      keys16, w1k, kb1, w2k, kb2, ke16);
  qe_enc_wmma<<<B_ * (T1_ / 64), 256, 0, stream>>>(
      qin16, w1q, qb1, w2q, qb2, w3q, qb3, qe16);

  // norms
  rownorm_kernel<<<(B_ * T2_ + 255) / 256, 256, 0, stream>>>(ke16, k2, B_ * T2_);
  rownorm_kernel<<<(B_ * T1_ + 255) / 256, 256, 0, stream>>>(qe16, q2, B_ * T1_);

  // fused attention
  attn_kernel<<<B_ * (T1_ / 16), 256, 0, stream>>>(
      qe16, ke16, q2, k2, prior, mask, attn_out, attn_lp);
}
